// Noisy_Conditioner_Q_Former_3358664426227
// MI455X (gfx1250) — matrix operand order repaired and checked by reading, not compiled, not run on hardware
//
#include <hip/hip_runtime.h>
#include <math.h>

// ---------------------------------------------------------------------------
// Model constants (from reference)
// ---------------------------------------------------------------------------
#define DIM      1024
#define NUM_Q    512
#define NFRAMES  64
#define NPATCH   64          // 8x8 grid per frame
#define NB       2           // batch
#define SEQ      4098        // 2 + 64*64 tokens
#define KVLEN    4610        // SEQ + NUM_Q
#define KVPAD    4640        // padded to multiple of 32
#define HEADS    16
#define DH       88
#define DHP      96          // head dim padded to multiple of 32
#define INNER    1408        // HEADS*DH
#define INNERP   1536        // HEADS*DHP
#define FF       4096
#define PK       256         // patch GEMM K = 4*8*8

typedef __bf16 bf16;
typedef __attribute__((ext_vector_type(16))) __bf16 bf16x16;
typedef __attribute__((ext_vector_type(8)))  float  f32x8;

union FragAB { bf16x16 v16; int4 q[2]; };
union FragC  { f32x8 v; float f[8]; };

__device__ __forceinline__ bf16 f2bf(float f) {
    unsigned u = __builtin_bit_cast(unsigned, f);
    unsigned r = u + 0x7fffu + ((u >> 16) & 1u);   // RNE
    unsigned short h = (unsigned short)(r >> 16);
    return __builtin_bit_cast(bf16, h);
}

// ---------------------------------------------------------------------------
// Generic strided-batched bf16 WMMA GEMM: C = A[M,K] * W[N,K]^T (+epilogue)
// Fragment layouts per CDNA5 ISA (16-bit A 16x32 / B 32x16).
// OOB fragment rows are CLAMPED (not zeroed): each output element (row,col)
// depends only on A-row `row` and W-row `col`, and OOB outputs are masked at
// store, so clamped garbage never reaches memory. Inner loop is exec-mask
// free and explicitly software-pipelined (copy-free ping-pong buffers).
// ---------------------------------------------------------------------------
enum { F_BIAS = 1, F_GELU = 2, F_RES = 4, F_BF16 = 8, F_TRANS = 16 };

template<int EPI>
__global__ __launch_bounds__(256) void gemm_bf16_wmma(
    const bf16* __restrict__ A, long long sA, int lda,
    const bf16* __restrict__ W, long long sW, int ldw,
    void* __restrict__ C, long long sC, int ldc,
    const float* __restrict__ bias,
    const float* __restrict__ R, long long sR, int ldr,
    int M, int N, int K, float scale)
{
    const bf16* Ab = A + (long long)blockIdx.z * sA;
    const bf16* Wb = W + (long long)blockIdx.z * sW;
    const int m0   = blockIdx.y * 128;
    const int n0   = blockIdx.x * 128;
    const int lane = threadIdx.x & 31;
    const int wid  = threadIdx.x >> 5;
    const int wm   = (wid >> 1) * 32;              // 4 waves along M
    const int wn   = (wid & 1)  * 64;              // 2 waves along N
    const int l15  = lane & 15;
    const int half = lane >> 4;

    // Row-clamped per-fragment base pointers (hoisted out of the k-loop).
    const bf16* pa[2];
    #pragma unroll
    for (int i = 0; i < 2; ++i) {
        int row = m0 + wm + 16 * i + l15;
        if (row >= M) row = M - 1;
        pa[i] = Ab + (long long)row * lda + half * 8;    // A: halves 0-7 @k, 8-15 @k+16
    }
    const bf16* pw[4];
    #pragma unroll
    for (int j = 0; j < 4; ++j) {
        int row = n0 + wn + 16 * j + l15;
        if (row >= N) row = N - 1;
        pw[j] = Wb + (long long)row * ldw + half * 8;    // B: halves 0-7 @k, 8-15 @k+16
    }

    FragC acc[2][4];
    #pragma unroll
    for (int i = 0; i < 2; ++i)
        #pragma unroll
        for (int j = 0; j < 4; ++j)
            #pragma unroll
            for (int e = 0; e < 8; ++e) acc[i][j].f[e] = 0.f;

    auto loadA = [&](FragAB fa[2], int kk) {
        #pragma unroll
        for (int i = 0; i < 2; ++i) {
            fa[i].q[0] = *(const int4*)(pa[i] + kk);
            fa[i].q[1] = *(const int4*)(pa[i] + kk + 16);
        }
    };
    auto loadW = [&](FragAB fw[4], int kk) {
        #pragma unroll
        for (int j = 0; j < 4; ++j) {
            fw[j].q[0] = *(const int4*)(pw[j] + kk);
            fw[j].q[1] = *(const int4*)(pw[j] + kk + 16);
        }
    };
    auto mma = [&](FragAB fa[2], FragAB fw[4]) {
        #pragma unroll
        for (int i = 0; i < 2; ++i)
            #pragma unroll
            for (int j = 0; j < 4; ++j)
                acc[i][j].v = __builtin_amdgcn_wmma_f32_16x16x32_bf16(
                    false, fa[i].v16, false, fw[j].v16, (short)0, acc[i][j].v, false, false);
    };

    // Copy-free ping-pong pipeline, unrolled by two 32-deep tiles.
    // Invariant at loop head: a0/w0 holds tile k0.
    FragAB a0[2], w0[4], a1[2], w1[4];
    loadA(a0, 0); loadW(w0, 0);
    int k0 = 0;
    for (; k0 + 64 <= K; k0 += 64) {
        loadA(a1, k0 + 32); loadW(w1, k0 + 32);
        __builtin_prefetch(pa[0] + k0 + 96, 0, 1);
        __builtin_prefetch(pw[0] + k0 + 96, 0, 1);
        mma(a0, w0);
        if (k0 + 64 < K) { loadA(a0, k0 + 64); loadW(w0, k0 + 64); }
        mma(a1, w1);
    }
    if (k0 < K) mma(a0, w0);   // odd tile count: last tile already resident

    const int colBase = n0 + wn + l15;
    const int rowBase = m0 + wm + (half << 3);
    for (int i = 0; i < 2; ++i) {
        for (int r = 0; r < 8; ++r) {
            int row = rowBase + 16 * i + r;
            if (row >= M) continue;
            for (int j = 0; j < 4; ++j) {
                int col = colBase + 16 * j;
                if (col >= N) continue;
                float v = acc[i][j].f[r];
                if (EPI & F_BIAS) v += bias[col];
                v *= scale;
                if (EPI & F_GELU) v = 0.5f * v * (1.f + erff(v * 0.70710678118654752f));
                if (EPI & F_RES)  v += R[(long long)blockIdx.z * sR + (long long)row * ldr + col];
                long long off = (EPI & F_TRANS) ? ((long long)col * ldc + row)
                                                : ((long long)row * ldc + col);
                long long idx = (long long)blockIdx.z * sC + off;
                if (EPI & F_BF16) ((bf16*)C)[idx] = f2bf(v);
                else              ((float*)C)[idx] = v;
            }
        }
    }
}

// ---------------------------------------------------------------------------
// LayerNorm over 1024 dims; one block per row
// ---------------------------------------------------------------------------
template<bool BF16OUT>
__global__ __launch_bounds__(256) void layernorm_k(
    const float* __restrict__ in, long long inStride,
    const float* __restrict__ g, const float* __restrict__ bb,
    void* __restrict__ out, long long outStride)
{
    __shared__ float sd[256];
    const int row = blockIdx.x, b = blockIdx.y, tid = threadIdx.x;
    const float* p = in + (long long)b * inStride + (long long)row * DIM;
    float v0 = p[tid], v1 = p[tid + 256], v2 = p[tid + 512], v3 = p[tid + 768];
    float s = v0 + v1 + v2 + v3;
    sd[tid] = s; __syncthreads();
    for (int st = 128; st > 0; st >>= 1) { if (tid < st) sd[tid] += sd[tid + st]; __syncthreads(); }
    float mean = sd[0] * (1.f / DIM); __syncthreads();
    float d0 = v0 - mean, d1 = v1 - mean, d2 = v2 - mean, d3 = v3 - mean;
    sd[tid] = d0*d0 + d1*d1 + d2*d2 + d3*d3; __syncthreads();
    for (int st = 128; st > 0; st >>= 1) { if (tid < st) sd[tid] += sd[tid + st]; __syncthreads(); }
    float rstd = rsqrtf(sd[0] * (1.f / DIM) + 1e-5f);
    long long base = (long long)b * outStride + (long long)row * DIM;
    float dv[4] = {d0, d1, d2, d3};
    for (int t = 0; t < 4; ++t) {
        int i = tid + 256 * t;
        float o = dv[t] * rstd * g[i] + bb[i];
        if (BF16OUT) ((bf16*)out)[base + i] = f2bf(o);
        else         ((float*)out)[base + i] = o;
    }
}

// ---------------------------------------------------------------------------
// Row softmax, fp32 in / bf16 out written in-place (pitch trick)
// ---------------------------------------------------------------------------
__global__ __launch_bounds__(256) void softmax_rows_k(float* __restrict__ sim, int valid, int width)
{
    __shared__ float buf[KVPAD];
    __shared__ float sd[256];
    const long long row = blockIdx.x;
    float* p = sim + row * (long long)width;
    const int tid = threadIdx.x;
    float mx = -3.4e38f;
    for (int i = tid; i < width; i += 256) {
        float v = (i < valid) ? p[i] : -3.4e38f;
        buf[i] = v; mx = fmaxf(mx, v);
    }
    sd[tid] = mx; __syncthreads();
    for (int st = 128; st > 0; st >>= 1) { if (tid < st) sd[tid] = fmaxf(sd[tid], sd[tid + st]); __syncthreads(); }
    mx = sd[0]; __syncthreads();
    float sum = 0.f;
    for (int i = tid; i < width; i += 256) {
        float e = (i < valid) ? __expf(buf[i] - mx) : 0.f;
        buf[i] = e; sum += e;
    }
    sd[tid] = sum; __syncthreads();
    for (int st = 128; st > 0; st >>= 1) { if (tid < st) sd[tid] += sd[tid + st]; __syncthreads(); }
    float inv = 1.f / sd[0];
    bf16* outp = (bf16*)sim + row * (long long)(2 * width);  // same bytes, bf16 pitch
    for (int i = tid; i < width; i += 256) outp[i] = f2bf(buf[i] * inv);
}

// ---------------------------------------------------------------------------
// Small prep kernels
// ---------------------------------------------------------------------------
__global__ void cvt_bf16_k(const float* __restrict__ in, bf16* __restrict__ out, int n) {
    int i = blockIdx.x * 256 + threadIdx.x;
    if (i < n) out[i] = f2bf(in[i]);
}

// x[2,64,4,64,64] -> A_patch[8192, 256]  (K order = c, ph, pw)
__global__ void im2col_k(const float* __restrict__ x, bf16* __restrict__ out) {
    int i = blockIdx.x * 256 + threadIdx.x;
    if (i >= 8192 * PK) return;
    int m = i / PK, k = i % PK;
    int bf = m / NPATCH, pch = m % NPATCH;
    int gh = pch >> 3, gw = pch & 7;
    int c = k / 64, r = k % 64, ph = r >> 3, pw = r & 7;
    out[i] = f2bf(x[(((long long)bf * 4 + c) * 64 + gh * 8 + ph) * 64 + gw * 8 + pw]);
}

__device__ __forceinline__ float posemb(int p, int d) {
    int gh = p >> 3, gw = p & 7;
    float coord; int dd;
    if (d < 512) { coord = (float)gh; dd = d; }
    else         { coord = (float)gw; dd = d - 512; }
    int i = dd & 255;
    float omega = __expf(-(float)i * (9.210340371976184f / 256.f));  // 10000^(-i/256)
    float a = coord * omega;
    return (dd < 256) ? __sinf(a) : __cosf(a);
}

// seq rows 2.. already hold z+conv_b; add pos + frame_pos, write tokens 0,1
__global__ void finish_seq_k(float* __restrict__ seq,
                             const float* __restrict__ time_emb, const float* __restrict__ frame_emb,
                             const float* __restrict__ frame_pos,
                             const int* __restrict__ tsteps, const int* __restrict__ fidx) {
    long long i = (long long)blockIdx.x * 256 + threadIdx.x;
    if (i >= (long long)NB * SEQ * DIM) return;
    int d = (int)(i % DIM);
    int t = (int)((i / DIM) % SEQ);
    int b = (int)(i / ((long long)DIM * SEQ));
    if (t == 0)      seq[i] = time_emb[(long long)tsteps[b] * DIM + d];
    else if (t == 1) seq[i] = frame_emb[(long long)fidx[b] * DIM + d];
    else {
        int tok = t - 2, fr = tok >> 6, pp = tok & 63;
        seq[i] += posemb(pp, d) + frame_pos[fr * DIM + d];
    }
}

__global__ void init_lat_k(const float* __restrict__ latents, float* __restrict__ lat) {
    long long i = (long long)blockIdx.x * 256 + threadIdx.x;
    if (i < (long long)NB * NUM_Q * DIM) lat[i] = latents[i % ((long long)NUM_Q * DIM)];
}

// src [DIM, src_ld] cols [col_off .. col_off+1408) -> dst [INNERP, DIM], head-dim padded
__global__ void headpadT_k(const float* __restrict__ src, int src_ld, int col_off,
                           bf16* __restrict__ dst) {
    int i = blockIdx.x * 256 + threadIdx.x;
    if (i >= INNERP * DIM) return;
    int n = i / DIM, k = i % DIM;
    int h = n / DHP, d = n % DHP;
    dst[i] = (d < DH) ? f2bf(src[(long long)k * src_ld + col_off + h * DH + d]) : f2bf(0.f);
}

__global__ void padbias_k(const float* __restrict__ src, int off, float* __restrict__ dst) {
    int n = blockIdx.x * 256 + threadIdx.x;
    if (n >= INNERP) return;
    int h = n / DHP, d = n % DHP;
    dst[n] = (d < DH) ? src[off + h * DH + d] : 0.f;
}

// wo [INNER, DIM] -> dst [DIM, INNERP] (K = padded head layout)
__global__ void woT_pad_k(const float* __restrict__ wo, bf16* __restrict__ dst) {
    int i = blockIdx.x * 256 + threadIdx.x;
    if (i >= DIM * INNERP) return;
    int n = i / INNERP, kk = i % INNERP;
    int h = kk / DHP, d = kk % DHP;
    dst[i] = (d < DH) ? f2bf(wo[(long long)(h * DH + d) * DIM + n]) : f2bf(0.f);
}

// ---------------------------------------------------------------------------
// Tiled transpose via async global->LDS staging (CDNA5 ASYNCcnt path).
// src [rows, cols] fp32 -> dst [cols, rows] bf16.  rows, cols multiples of 64.
// Staging loads use GLOBAL_LOAD_ASYNC_TO_LDS_B128 (GV addressing), completed
// with S_WAIT_ASYNCCNT 0 before the workgroup barrier.
// ---------------------------------------------------------------------------
__global__ __launch_bounds__(256) void transposeT_k(const float* __restrict__ src,
                                                    int rows, int cols,
                                                    bf16* __restrict__ dst) {
    __shared__ float tile[64][68];                 // pitch 68 floats: 16B-aligned rows
    const int r0 = blockIdx.y * 64, c0 = blockIdx.x * 64;
    const int tid = threadIdx.x;
    #pragma unroll
    for (int t = 0; t < 4; ++t) {
        int ch = tid + 256 * t;                    // 1024 x 16B chunks
        int rr = ch >> 4, cc = (ch & 15) << 2;
        unsigned lds_off = (unsigned)(size_t)&tile[rr][cc];
        const float* g = src + (long long)(r0 + rr) * cols + (c0 + cc);
        asm volatile("global_load_async_to_lds_b128 %0, %1, off"
                     :: "v"(lds_off), "v"(g) : "memory");
    }
    asm volatile("s_wait_asynccnt 0x0" ::: "memory");
    __syncthreads();
    #pragma unroll
    for (int t = 0; t < 16; ++t) {
        int idx = tid + 256 * t;                   // 4096 outputs per block
        int cc = idx >> 6, rr = idx & 63;
        dst[(long long)(c0 + cc) * rows + (r0 + rr)] = f2bf(tile[rr][cc]);
    }
}

// ---------------------------------------------------------------------------
// Host orchestration
// ---------------------------------------------------------------------------
#define GEMM(FLAGS, A_, sA_, lda_, W_, sW_, ldw_, C_, sC_, ldc_, bias_, R_, sR_, ldr_, M_, N_, K_, scale_, nb_) \
    gemm_bf16_wmma<(FLAGS)><<<dim3(((N_) + 127) / 128, ((M_) + 127) / 128, (nb_)), 256, 0, stream>>>( \
        (A_), (sA_), (lda_), (W_), (sW_), (ldw_), (void*)(C_), (sC_), (ldc_), (bias_), (R_), (sR_), (ldr_), \
        (M_), (N_), (K_), (scale_))

extern "C" void kernel_launch(void* const* d_in, const int* in_sizes, int n_in,
                              void* d_out, int out_size, void* d_ws, size_t ws_size,
                              hipStream_t stream) {
    (void)in_sizes; (void)n_in; (void)out_size; (void)ws_size;

    // ---- unpack inputs (setup_inputs() dict insertion order, depth-first) ----
    int ii = 0;
    const float* x          = (const float*)d_in[ii++];
    const float* conv_w     = (const float*)d_in[ii++];
    const float* conv_b     = (const float*)d_in[ii++];
    const float* latents    = (const float*)d_in[ii++];
    const float* frame_pos  = (const float*)d_in[ii++];
    const float* frame_emb  = (const float*)d_in[ii++];
    const float* time_emb   = (const float*)d_in[ii++];
    const float* norm_g     = (const float*)d_in[ii++];
    const float* norm_b     = (const float*)d_in[ii++];
    struct LayerW { const float *ln_m_g,*ln_m_b,*ln_l_g,*ln_l_b,*wq,*bq,*wkv,*bkv,*wo,*bo,*ln_g,*ln_b,*w1,*w2; } L[2];
    for (int l = 0; l < 2; ++l) {
        L[l].ln_m_g = (const float*)d_in[ii++]; L[l].ln_m_b = (const float*)d_in[ii++];
        L[l].ln_l_g = (const float*)d_in[ii++]; L[l].ln_l_b = (const float*)d_in[ii++];
        L[l].wq  = (const float*)d_in[ii++];    L[l].bq  = (const float*)d_in[ii++];
        L[l].wkv = (const float*)d_in[ii++];    L[l].bkv = (const float*)d_in[ii++];
        L[l].wo  = (const float*)d_in[ii++];    L[l].bo  = (const float*)d_in[ii++];
        L[l].ln_g = (const float*)d_in[ii++];   L[l].ln_b = (const float*)d_in[ii++];
        L[l].w1  = (const float*)d_in[ii++];    L[l].w2  = (const float*)d_in[ii++];
    }
    const int* tsteps = (const int*)d_in[ii++];
    const int* fidx   = (const int*)d_in[ii++];

    // ---- workspace carve ----
    char* wp = (char*)d_ws;
    auto carve = [&](size_t bytes) -> void* {
        void* r = (void*)wp; wp += (bytes + 255) & ~(size_t)255; return r;
    };
    bf16*  apatch = (bf16*) carve((size_t)8192 * PK * 2);
    bf16*  convWb = (bf16*) carve((size_t)DIM * PK * 2);
    float* seq    = (float*)carve((size_t)NB * SEQ * DIM * 4);
    bf16*  kvin   = (bf16*) carve((size_t)NB * KVLEN * DIM * 2);
    float* lat    = (float*)carve((size_t)NB * NUM_Q * DIM * 4);
    bf16*  wqT    = (bf16*) carve((size_t)INNERP * DIM * 2);
    bf16*  wkT    = (bf16*) carve((size_t)INNERP * DIM * 2);
    bf16*  wvT    = (bf16*) carve((size_t)INNERP * DIM * 2);
    float* bqp    = (float*)carve((size_t)INNERP * 4);
    float* bkp    = (float*)carve((size_t)INNERP * 4);
    float* bvp    = (float*)carve((size_t)INNERP * 4);
    bf16*  woT    = (bf16*) carve((size_t)DIM * INNERP * 2);
    bf16*  w1T    = (bf16*) carve((size_t)FF * DIM * 2);
    bf16*  w2T    = (bf16*) carve((size_t)DIM * FF * 2);
    bf16*  qp     = (bf16*) carve((size_t)NB * NUM_Q * INNERP * 2);
    bf16*  kp     = (bf16*) carve((size_t)NB * KVLEN * INNERP * 2);
    bf16*  vT     = (bf16*) carve((size_t)NB * INNERP * KVPAD * 2);
    float* sim    = (float*)carve((size_t)HEADS * NUM_Q * KVPAD * 4);   // reused per batch
    bf16*  ob     = (bf16*) carve((size_t)NB * NUM_Q * INNERP * 2);
    bf16*  lnl    = (bf16*) carve((size_t)NB * NUM_Q * DIM * 2);
    bf16*  hb     = (bf16*) carve((size_t)NB * NUM_Q * FF * 2);

    auto blocks = [](long long n) { return (unsigned)((n + 255) / 256); };

    // zero Vt pad columns once (pad cols never written afterwards)
    hipMemsetAsync(vT, 0, (size_t)NB * INNERP * KVPAD * 2, stream);

    // ---- patch embed: im2col + GEMM into seq rows 2.. ----
    cvt_bf16_k<<<blocks(DIM * PK), 256, 0, stream>>>(conv_w, convWb, DIM * PK);
    im2col_k<<<blocks((long long)8192 * PK), 256, 0, stream>>>(x, apatch);
    GEMM(F_BIAS, apatch, (long long)4096 * PK, PK, convWb, 0LL, PK,
         seq + 2 * DIM, (long long)SEQ * DIM, DIM, conv_b, (const float*)nullptr, 0LL, 0,
         4096, DIM, PK, 1.f, NB);
    finish_seq_k<<<blocks((long long)NB * SEQ * DIM), 256, 0, stream>>>(
        seq, time_emb, frame_emb, frame_pos, tsteps, fidx);
    init_lat_k<<<blocks((long long)NB * NUM_Q * DIM), 256, 0, stream>>>(latents, lat);

    const float qscale = 0.10660035817780522f;  // 88^-0.5

    for (int l = 0; l < 2; ++l) {
        const LayerW& Lw = L[l];
        // -- weight prep (padded, transposed, bf16) --
        headpadT_k<<<blocks(INNERP * DIM), 256, 0, stream>>>(Lw.wq,  INNER,     0,     wqT);
        headpadT_k<<<blocks(INNERP * DIM), 256, 0, stream>>>(Lw.wkv, 2 * INNER, 0,     wkT);
        headpadT_k<<<blocks(INNERP * DIM), 256, 0, stream>>>(Lw.wkv, 2 * INNER, INNER, wvT);
        padbias_k<<<blocks(INNERP), 256, 0, stream>>>(Lw.bq,  0,     bqp);
        padbias_k<<<blocks(INNERP), 256, 0, stream>>>(Lw.bkv, 0,     bkp);
        padbias_k<<<blocks(INNERP), 256, 0, stream>>>(Lw.bkv, INNER, bvp);
        woT_pad_k<<<blocks(DIM * INNERP), 256, 0, stream>>>(Lw.wo, woT);
        transposeT_k<<<dim3(FF / 64, DIM / 64), 256, 0, stream>>>(Lw.w1, DIM, FF, w1T);
        transposeT_k<<<dim3(DIM / 64, FF / 64), 256, 0, stream>>>(Lw.w2, FF, DIM, w2T);

        // -- layernorms into kv_in (bf16): seq rows then latent rows --
        layernorm_k<true><<<dim3(SEQ, NB), 256, 0, stream>>>(
            seq, (long long)SEQ * DIM, Lw.ln_m_g, Lw.ln_m_b, kvin, (long long)KVLEN * DIM);
        layernorm_k<true><<<dim3(NUM_Q, NB), 256, 0, stream>>>(
            lat, (long long)NUM_Q * DIM, Lw.ln_l_g, Lw.ln_l_b,
            kvin + (long long)SEQ * DIM, (long long)KVLEN * DIM);

        // -- q/k/v projections (head-padded) --
        GEMM(F_BIAS | F_BF16, kvin + (long long)SEQ * DIM, (long long)KVLEN * DIM, DIM,
             wqT, 0LL, DIM, qp, (long long)NUM_Q * INNERP, INNERP, bqp,
             (const float*)nullptr, 0LL, 0, NUM_Q, INNERP, DIM, qscale, NB);
        GEMM(F_BIAS | F_BF16, kvin, (long long)KVLEN * DIM, DIM,
             wkT, 0LL, DIM, kp, (long long)KVLEN * INNERP, INNERP, bkp,
             (const float*)nullptr, 0LL, 0, KVLEN, INNERP, DIM, 1.f, NB);
        GEMM(F_BIAS | F_BF16 | F_TRANS, kvin, (long long)KVLEN * DIM, DIM,
             wvT, 0LL, DIM, vT, (long long)INNERP * KVPAD, KVPAD, bvp,
             (const float*)nullptr, 0LL, 0, KVLEN, INNERP, DIM, 1.f, NB);

        // -- attention per batch (heads batched in z) --
        for (int b = 0; b < NB; ++b) {
            GEMM(0, qp + (long long)b * NUM_Q * INNERP, (long long)DHP, INNERP,
                 kp + (long long)b * KVLEN * INNERP, (long long)DHP, INNERP,
                 sim, (long long)NUM_Q * KVPAD, KVPAD, (const float*)nullptr,
                 (const float*)nullptr, 0LL, 0, NUM_Q, KVLEN, DHP, 1.f, HEADS);
            softmax_rows_k<<<HEADS * NUM_Q, 256, 0, stream>>>(sim, KVLEN, KVPAD);
            // attn (bf16, pitch 2*KVPAD) x V^T -> o (bf16, head-padded)
            GEMM(F_BF16, (const bf16*)sim, (long long)NUM_Q * 2 * KVPAD, 2 * KVPAD,
                 vT + (long long)b * INNERP * KVPAD, (long long)DHP * KVPAD, KVPAD,
                 ob + (long long)b * NUM_Q * INNERP, (long long)DHP, INNERP,
                 (const float*)nullptr, (const float*)nullptr, 0LL, 0,
                 NUM_Q, DHP, KVPAD, 1.f, HEADS);
        }

        // -- output projection + residual: lat = o @ wo + bo + lat --
        GEMM(F_BIAS | F_RES, ob, 0LL, INNERP, woT, 0LL, INNERP,
             lat, 0LL, DIM, Lw.bo, lat, 0LL, DIM,
             NB * NUM_Q, DIM, INNERP, 1.f, 1);

        // -- FF block: lat = gelu(ln(lat) @ w1) @ w2 + lat --
        layernorm_k<true><<<dim3(NB * NUM_Q, 1), 256, 0, stream>>>(
            lat, 0LL, Lw.ln_g, Lw.ln_b, lnl, 0LL);
        GEMM(F_GELU | F_BF16, lnl, 0LL, DIM, w1T, 0LL, DIM,
             hb, 0LL, FF, (const float*)nullptr, (const float*)nullptr, 0LL, 0,
             NB * NUM_Q, FF, DIM, 1.f, 1);
        GEMM(F_RES, hb, 0LL, FF, w2T, 0LL, FF,
             lat, 0LL, DIM, (const float*)nullptr, lat, 0LL, DIM,
             NB * NUM_Q, DIM, FF, 1.f, 1);
    }

    // -- final layernorm -> d_out (fp32) --
    layernorm_k<false><<<dim3(NB * NUM_Q, 1), 256, 0, stream>>>(
        lat, 0LL, norm_g, norm_b, (float*)d_out, 0LL);
}
